// LDABBVI_27127013441905
// MI455X (gfx1250) — matrix operand order repaired and checked by reading, not compiled, not run on hardware
//
#include <hip/hip_runtime.h>
#include <hip/hip_bf16.h>
#include <math.h>

#define D_DOCS 512
#define K_TOP  128
#define V_VOC  8000
#define S_SMP  8

typedef __attribute__((ext_vector_type(16))) __bf16 v16bf;
typedef __attribute__((ext_vector_type(8)))  __bf16 v8bf;
typedef __attribute__((ext_vector_type(8)))  float  v8f;

union V16U { v16bf v; v8bf h[2]; };

static __device__ __forceinline__ unsigned hash32(unsigned x){
  x ^= x >> 16; x *= 0x7feb352du; x ^= x >> 15; x *= 0x846ca68bu; x ^= x >> 16;
  return x;
}
static __device__ __forceinline__ float u01(unsigned h){
  return (float)(h >> 8) * (1.0f/16777216.0f) + (0.5f/16777216.0f);
}
static __device__ __forceinline__ __bf16 f2bf(float f){
  union { float f; unsigned u; } in; in.f = f;
  unsigned r = (in.u + 0x7fffu + ((in.u >> 16) & 1u)) >> 16;
  union { unsigned short s; __bf16 b; } out; out.s = (unsigned short)r;
  return out.b;
}

// CDNA5 async copy: global -> LDS, tracked by ASYNCcnt (16 bytes per lane).
static __device__ __forceinline__ void async_copy_b128(unsigned lds_off, const void* gaddr){
  asm volatile("global_load_async_to_lds_b128 %0, %1, off"
               :: "v"(lds_off), "v"(gaddr) : "memory");
}
static __device__ __forceinline__ void wait_asynccnt0(){
  asm volatile("s_wait_asynccnt 0" ::: "memory");
}

// log-space Gamma(alpha) sample: Marsaglia-Tsang on alpha+1 plus log(u)/alpha boost.
// Deterministic in `key` (counter-based RNG) so it can be recomputed in pass 2.
static __device__ float log_gamma_sample(float alpha, unsigned key){
  float a = alpha + 1.0f;
  float d = a - (1.0f/3.0f);
  float c = rsqrtf(9.0f * d);
  float logdv = __logf(d);          // fallback if no acceptance (vanishingly rare)
  for (int t = 0; t < 24; ++t){
    unsigned b = key + (unsigned)t * 3u;
    float u1 = u01(hash32(b));
    float u2 = u01(hash32(b + 1u));
    float r  = sqrtf(-2.0f * __logf(u1));
    float n  = r * __cosf(6.2831853f * u2);
    float v1 = 1.0f + c * n;
    if (v1 <= 0.0f) continue;
    float v3 = v1 * v1 * v1;
    float u  = u01(hash32(b + 2u));
    if (__logf(u) < 0.5f*n*n + d - d*v3 + d*__logf(v3)){
      logdv = __logf(d * v3);
      break;
    }
  }
  float ub = u01(hash32(key + 1013u));
  return logdv + __logf(ub) / alpha;   // log of Gamma(alpha) draw
}

__global__ void lda_init(float* acc){
  if (threadIdx.x < 16) acc[threadIdx.x] = 0.0f;   // [0..7]=loglik, [8..15]=logq
}

// One block per (s,k) topic row: sample Dirichlet row in log space (online LSE),
// write bf16 TRANSPOSED topics [S][V][K], accumulate log q terms.
__global__ __launch_bounds__(256) void lda_sample_topics(
    const float* __restrict__ tlv, __bf16* __restrict__ topicsT, float* __restrict__ acc)
{
  const int tid  = threadIdx.x;
  const int rowg = blockIdx.x;               // [0, S*K)
  const int s = rowg / K_TOP;
  const int k = rowg % K_TOP;
  const float* lrow = tlv + (size_t)k * V_VOC;
  const unsigned seed = 0x9E3779B9u + (unsigned)rowg * 0x01000193u;

  float m = -__builtin_inff(), sum = 0.0f, lamsum = 0.0f, lgsum = 0.0f;
  for (int v = tid; v < V_VOC; v += 256){
    float lam = log1pf(__expf(lrow[v])) + 1e-6f;        // softplus + eps
    unsigned key = seed ^ ((unsigned)v * 0x9E3779B1u);
    float lg = log_gamma_sample(lam, key);
    lamsum += lam;
    lgsum  += lgammaf(lam);
    if (lg > m){ sum = sum * __expf(m - lg) + 1.0f; m = lg; }
    else         sum += __expf(lg - m);
  }
  __shared__ float sm[256], ss[256], sa[256], sb[256];
  sm[tid]=m; ss[tid]=sum; sa[tid]=lamsum; sb[tid]=lgsum;
  __syncthreads();
  for (int st = 128; st > 0; st >>= 1){
    if (tid < st){
      float m2 = sm[tid+st], c2 = ss[tid+st];
      if (c2 > 0.0f){
        if (m2 > sm[tid]){ ss[tid] = ss[tid]*__expf(sm[tid]-m2) + c2; sm[tid] = m2; }
        else              ss[tid] += c2 * __expf(m2 - sm[tid]);
      }
      sa[tid] += sa[tid+st]; sb[tid] += sb[tid+st];
    }
    __syncthreads();
  }
  const float lse  = sm[0] + __logf(ss[0]);
  const float lamS = sa[0], lgS = sb[0];
  __syncthreads();

  float q = 0.0f;
  for (int v = tid; v < V_VOC; v += 256){
    float lam = log1pf(__expf(lrow[v])) + 1e-6f;
    unsigned key = seed ^ ((unsigned)v * 0x9E3779B1u);
    float lg   = log_gamma_sample(lam, key);             // identical recompute
    float logx = lg - lse;
    q += (lam - 1.0f) * logx;
    topicsT[((size_t)s * V_VOC + v) * K_TOP + k] = f2bf(__expf(logx));
  }
  sa[tid] = q; __syncthreads();
  for (int st = 128; st > 0; st >>= 1){
    if (tid < st) sa[tid] += sa[tid+st];
    __syncthreads();
  }
  if (tid == 0) atomicAdd(&acc[8 + s], sa[0] + lgammaf(lamS) - lgS);
}

// One block (128 thr) per (s,d) doc row, K=128 elems, one per thread.
__global__ __launch_bounds__(128) void lda_sample_docs(
    const float* __restrict__ dlv, __bf16* __restrict__ docT, float* __restrict__ acc)
{
  const int tid  = threadIdx.x;
  const int rowg = blockIdx.x;               // [0, S*D)
  const int s = rowg / D_DOCS;
  const int d = rowg % D_DOCS;
  const float* grow = dlv + (size_t)d * K_TOP;
  const unsigned seed = 0x85EBCA6Bu + (unsigned)rowg * 0x01000193u;

  float gam = log1pf(__expf(grow[tid])) + 1e-6f;
  unsigned key = seed ^ ((unsigned)tid * 0x9E3779B1u);
  float lg = log_gamma_sample(gam, key);

  __shared__ float sm[128], ss[128], sa[128], sb[128];
  sm[tid]=lg; ss[tid]=1.0f; sa[tid]=gam; sb[tid]=lgammaf(gam);
  __syncthreads();
  for (int st = 64; st > 0; st >>= 1){
    if (tid < st){
      float m2 = sm[tid+st], c2 = ss[tid+st];
      if (m2 > sm[tid]){ ss[tid] = ss[tid]*__expf(sm[tid]-m2) + c2; sm[tid] = m2; }
      else              ss[tid] += c2 * __expf(m2 - sm[tid]);
      sa[tid] += sa[tid+st]; sb[tid] += sb[tid+st];
    }
    __syncthreads();
  }
  const float lse  = sm[0] + __logf(ss[0]);
  const float gamS = sa[0], lgS = sb[0];
  __syncthreads();

  float logx = lg - lse;
  docT[((size_t)s * D_DOCS + d) * K_TOP + tid] = f2bf(__expf(logx));
  sa[tid] = (gam - 1.0f) * logx;
  __syncthreads();
  for (int st = 64; st > 0; st >>= 1){
    if (tid < st) sa[tid] += sa[tid+st];
    __syncthreads();
  }
  if (tid == 0) atomicAdd(&acc[8 + s], sa[0] + lgammaf(gamS) - lgS);
}

// Fused batched GEMM (bf16 WMMA, f32 accum) + masked docs*log(wp) reduction.
// Block = 8 waves sharing one (s, 64-col V-strip): B strip (64x128 bf16 = 16KB)
// is staged ONCE into LDS via global_load_async_to_lds_b128 (ASYNCcnt), then each
// wave handles its own 16-row M-tile: A frags from global b128, B frags from
// ds_load_b128, 16 v_wmma_f32_16x16x32_bf16 per wave, branch-free fast-log epilogue.
__global__ __launch_bounds__(256) void lda_wmma_loglik(
    const __bf16* __restrict__ A,    // doc_topics bf16 [S][D][K]
    const __bf16* __restrict__ Bt,   // topics bf16 transposed [S][V][K]
    const int*    __restrict__ docs, // [D][V]
    float*        __restrict__ acc)  // loglik accumulators acc[0..7]
{
  __shared__ __bf16 ldsB[64 * K_TOP];          // 16 KB: [col][K]

  const int tid  = threadIdx.x;
  const int lane = tid & 31;
  const int wave = tid >> 5;

  const int VS  = V_VOC / 64;                  // 125 v-strips
  const int MTG = D_DOCS / 128;                // 4 groups of 8 M-tiles
  int vs  = blockIdx.x % VS;
  int t   = blockIdx.x / VS;
  int mtg = t % MTG;
  int s   = t / MTG;

  const __bf16* As = A  + (size_t)s * D_DOCS * K_TOP;
  const __bf16* Bs = Bt + ((size_t)s * V_VOC + (size_t)vs * 64) * K_TOP;

  // ---- stage B strip into LDS: 1024 x 16B chunks, 4 per thread, async ----
  const unsigned ldsBase = (unsigned)(size_t)(&ldsB[0]);
#pragma unroll
  for (int i = 0; i < 4; ++i){
    int c = tid + 256 * i;                     // chunk id: col = c/16, k8 = (c%16)*8
    async_copy_b128(ldsBase + (unsigned)c * 16u, (const void*)(Bs + (size_t)c * 8));
  }
  wait_asynccnt0();
  __syncthreads();

  // ---- per-wave M-tile ----
  const int mt    = mtg * 8 + wave;
  const int mbase = mt * 16;
  const int arow  = mbase + (lane & 15);
  const int akoff = (lane < 16) ? 0 : 8;       // A 16x32 layout: half-K split per lane group

  v16bf afr[4];
#pragma unroll
  for (int kb = 0; kb < 4; ++kb){
    const __bf16* p = As + (size_t)arow * K_TOP + kb*32 + akoff;
    V16U u; u.h[0] = *(const v8bf*)p; u.h[1] = *(const v8bf*)(p + 16);
    afr[kb] = u.v;
  }

  const int bkoff = (lane < 16) ? 0 : 8;       // B 32x16 layout: half-K split per lane group
  const int bcol0 = lane & 15;
  float lsum = 0.0f;
#pragma unroll
  for (int nt = 0; nt < 4; ++nt){
    v8f c = {};
#pragma unroll
    for (int kb = 0; kb < 4; ++kb){
      const __bf16* q = &ldsB[(nt*16 + bcol0) * K_TOP + kb*32 + bkoff];
      V16U u; u.h[0] = *(const v8bf*)q; u.h[1] = *(const v8bf*)(q + 16);
      c = __builtin_amdgcn_wmma_f32_16x16x32_bf16(
            false, afr[kb], false, u.v, (short)0, c, false, false);
    }
    // C layout: VGPR r, lanes 0-15 -> M=r, lanes 16-31 -> M=8+r; N = lane&15
    const int n    = vs*64 + nt*16 + bcol0;
    const int mofs = mbase + ((lane >> 4) << 3);
    // cnt==0 contributes exactly 0 (log is finite thanks to +1e-10): branch-free.
#pragma unroll
    for (int r = 0; r < 8; ++r){
      int cnt = docs[(size_t)(mofs + r) * V_VOC + n];
      lsum = fmaf((float)cnt, __logf(c[r] + 1e-10f), lsum);
    }
  }
#pragma unroll
  for (int off = 16; off > 0; off >>= 1)
    lsum += __shfl_down(lsum, off, 32);
  if (lane == 0) atomicAdd(&acc[s], lsum);
}

__global__ void lda_final(const float* __restrict__ acc, float* __restrict__ out){
  if (threadIdx.x == 0){
    // eta0 = alpha0 = 1 -> priors collapse to constants
    float cp = (float)K_TOP  * lgammaf((float)V_VOC)
             + (float)D_DOCS * lgammaf((float)K_TOP);
    float tot = 0.0f;
    for (int s = 0; s < S_SMP; ++s) tot += cp + acc[s] - acc[8 + s];
    out[0] = -(tot / (float)S_SMP);
  }
}

extern "C" void kernel_launch(void* const* d_in, const int* in_sizes, int n_in,
                              void* d_out, int out_size, void* d_ws, size_t ws_size,
                              hipStream_t stream){
  (void)in_sizes; (void)n_in; (void)out_size; (void)ws_size;
  const int*   docs = (const int*)d_in[0];
  const float* tlv  = (const float*)d_in[1];   // topic_log_var [K,V]
  const float* dlv  = (const float*)d_in[2];   // doc_log_var   [D,K]
  // d_in[3] = n_samples (compile-time S_SMP=8)

  float*  acc     = (float*)d_ws;                                        // 16 floats
  __bf16* topicsT = (__bf16*)((char*)d_ws + 256);                        // [S][V][K]
  __bf16* docT    = (__bf16*)((char*)d_ws + 256
                      + (size_t)S_SMP * V_VOC * K_TOP * sizeof(__bf16)); // [S][D][K]
  float* out = (float*)d_out;

  lda_init<<<1, 32, 0, stream>>>(acc);
  lda_sample_topics<<<S_SMP * K_TOP, 256, 0, stream>>>(tlv, topicsT, acc);
  lda_sample_docs  <<<S_SMP * D_DOCS, 128, 0, stream>>>(dlv, docT, acc);
  const int blocks = S_SMP * (D_DOCS/128) * (V_VOC/64);   // 8*4*125 = 4000 blocks
  lda_wmma_loglik<<<blocks, 256, 0, stream>>>(docT, topicsT, docs, acc);
  lda_final<<<1, 32, 0, stream>>>(acc, out);
}
